// AttentionBlock_34634616275502
// MI455X (gfx1250) — matrix operand order repaired and checked by reading, not compiled, not run on hardware
//
#include <hip/hip_runtime.h>
#include <cstddef>

typedef __attribute__((ext_vector_type(16))) __bf16 v16bf;
typedef __attribute__((ext_vector_type(8)))  __bf16 v8bf;
typedef __attribute__((ext_vector_type(8)))  float  v8f;

union FragA { v16bf v; v8bf h[2]; };

constexpr int Bb = 8, Cc = 256, Nn = 4096, C8 = 32;
constexpr float LOG2E = 1.4426950408889634f;

// ---- canonicalize-free IEEE max via direct ISA ops ----
__device__ __forceinline__ float amax(float a, float b) {
  float d; asm("v_max_num_f32 %0, %1, %2" : "=v"(d) : "v"(a), "v"(b)); return d;
}
__device__ __forceinline__ float amax3(float a, float b, float c) {
  float d; asm("v_max3_num_f32 %0, %1, %2, %3" : "=v"(d) : "v"(a), "v"(b), "v"(c)); return d;
}

// ---- fused-DPP 16-lane butterfly reductions (1 instruction per step) ----
// dst = op(dpp(src0), src1); all permutes stay inside the DPP row (16 lanes),
// which is exactly our N-group. CDNA5 interlocks VALU->DPP deps in hardware.
__device__ __forceinline__ float rowmax16(float x) {
  asm("v_max_num_f32_dpp %0, %0, %0 quad_perm:[1,0,3,2] row_mask:0xf bank_mask:0xf bound_ctrl:1" : "+v"(x));
  asm("v_max_num_f32_dpp %0, %0, %0 quad_perm:[2,3,0,1] row_mask:0xf bank_mask:0xf bound_ctrl:1" : "+v"(x));
  asm("v_max_num_f32_dpp %0, %0, %0 row_half_mirror row_mask:0xf bank_mask:0xf bound_ctrl:1" : "+v"(x));
  asm("v_max_num_f32_dpp %0, %0, %0 row_mirror row_mask:0xf bank_mask:0xf bound_ctrl:1" : "+v"(x));
  return x;
}
__device__ __forceinline__ float rowsum16(float x) {
  asm("v_add_f32_dpp %0, %0, %0 quad_perm:[1,0,3,2] row_mask:0xf bank_mask:0xf bound_ctrl:1" : "+v"(x));
  asm("v_add_f32_dpp %0, %0, %0 quad_perm:[2,3,0,1] row_mask:0xf bank_mask:0xf bound_ctrl:1" : "+v"(x));
  asm("v_add_f32_dpp %0, %0, %0 row_half_mirror row_mask:0xf bank_mask:0xf bound_ctrl:1" : "+v"(x));
  asm("v_add_f32_dpp %0, %0, %0 row_mirror row_mask:0xf bank_mask:0xf bound_ctrl:1" : "+v"(x));
  return x;
}

// ---------------------------------------------------------------------------
// Phase 1: q,k,v projections (1x1 conv == GEMM), native-bf16 outputs in
// WMMA-fragment-friendly layouts. Q is pre-scaled by log2(e) so softmax can
// use raw v_exp_f32 (base-2) with equivalent results.
//   qT [B][N][32], kT [B][N][32], vB [B][C][N]
// ---------------------------------------------------------------------------
__global__ __launch_bounds__(256)
void qkv_bf16_kernel(const float* __restrict__ x,
                     const float* __restrict__ Wq, const float* __restrict__ bq,
                     const float* __restrict__ Wk, const float* __restrict__ bk,
                     const float* __restrict__ Wv, const float* __restrict__ bv,
                     __bf16* __restrict__ qT,
                     __bf16* __restrict__ kT,
                     __bf16* __restrict__ vB) {
  __shared__ float xl[Cc * 32];                         // 32 KB tile of x
  const int tid = threadIdx.x;
  const int b  = blockIdx.y;
  const int n0 = blockIdx.x * 32;

  for (int idx = tid; idx < Cc * 32; idx += 256) {
    int c = idx >> 5, n = idx & 31;
    xl[idx] = x[((size_t)(b * Cc + c)) * Nn + n0 + n];
  }
  __syncthreads();

  for (int e = tid; e < C8 * 32; e += 256) {
    int o = e >> 5, n = e & 31;
    float aq = bq[o], ak = bk[o];
#pragma unroll 4
    for (int c = 0; c < Cc; ++c) {
      float xv = xl[c * 32 + n];
      aq = fmaf(Wq[o * Cc + c], xv, aq);
      ak = fmaf(Wk[o * Cc + c], xv, ak);
    }
    size_t off = ((size_t)(b * Nn + n0 + n)) * C8 + o;
    qT[off] = (__bf16)(aq * LOG2E);                    // fold log2(e) into Q
    kT[off] = (__bf16)ak;
  }

  for (int e = tid; e < Cc * 32; e += 256) {
    int o = e >> 5, n = e & 31;
    float a = bv[o];
#pragma unroll 4
    for (int c = 0; c < Cc; ++c)
      a = fmaf(Wv[o * Cc + c], xl[c * 32 + n], a);
    vB[((size_t)(b * Cc + o)) * Nn + n0 + n] = (__bf16)a;
  }
}

// ---------------------------------------------------------------------------
// Phase 2: fused flash attention. Block = 64 query rows of one batch.
// 8 waves: wave w -> i-subtile (w>>1)*16, channel half (w&1)*128.
// j-tile = 64: 4 QK WMMAs + online softmax (fused-DPP butterflies) + 16 PV WMMAs.
// ---------------------------------------------------------------------------
__global__ __launch_bounds__(256)
void attn_flash_kernel(const __bf16* __restrict__ qT,
                       const __bf16* __restrict__ kT,
                       const __bf16* __restrict__ vB,
                       const float* __restrict__ x,
                       const float* __restrict__ gamma,
                       float* __restrict__ out) {
  constexpr int OTP = 65;                               // bank-conflict pad
  __shared__ __align__(32) __bf16 pbuf[8 * 16 * 64];    // per-wave P (16 KB)
  __shared__ float ot[Cc * OTP];                        // output transpose (~65 KB)

  const int tid  = threadIdx.x;
  const int w    = tid >> 5, lane = tid & 31;
  const int wi   = w >> 1,   ch   = w & 1;
  const int m    = lane & 15, hi  = lane >> 4;
  const int b    = blockIdx.y;
  const int i0   = blockIdx.x * 64;

  // A-fragment of Q (contiguous k order, same as the K B-frag: lane<16 K{0..15}, lane>=16 K{16..31})
  const __bf16* qrow = qT + ((size_t)(b * Nn + i0 + wi * 16 + m)) * C8;
  FragA aq;
  aq.v = *reinterpret_cast<const v16bf*>(qrow + hi * 16);

  v8f acc[8];
  v8f zc;
  float mrun[8], lrun[8];
#pragma unroll
  for (int r = 0; r < 8; ++r) {
    zc[r] = 0.0f; mrun[r] = -3.0e38f; lrun[r] = 0.0f;
#pragma unroll
    for (int t = 0; t < 8; ++t) acc[t][r] = 0.0f;
  }

  __bf16* pmine = pbuf + w * (16 * 64);
  const __bf16* kbB   = kT + ((size_t)(b * Nn)) * C8;
  const __bf16* vcol0 = vB + ((size_t)(b * Cc + ch * 128 + m)) * Nn;

  for (int j0 = 0; j0 < Nn; j0 += 64) {
    // ---- S = Qt x K over 64 columns (4 WMMAs) ----
    v8f s[4];
#pragma unroll
    for (int u = 0; u < 4; ++u) {
      v16bf bk = *reinterpret_cast<const v16bf*>(
          kbB + (size_t)(j0 + u * 16 + m) * C8 + hi * 16);
      s[u] = __builtin_amdgcn_wmma_f32_16x16x32_bf16(false, aq.v, false, bk,
                                                     (short)0, zc, false, false);
    }

    // ---- online softmax over the 16x64 tile (base-2 domain) ----
    float alpha[8];
#pragma unroll
    for (int r = 0; r < 8; ++r) {
      float mloc = rowmax16(amax(amax3(s[0][r], s[1][r], s[2][r]), s[3][r]));
      float mnew = amax(mrun[r], mloc);
      float a  = __builtin_amdgcn_exp2f(mrun[r] - mnew);
      float p0 = __builtin_amdgcn_exp2f(s[0][r] - mnew);
      float p1 = __builtin_amdgcn_exp2f(s[1][r] - mnew);
      float p2 = __builtin_amdgcn_exp2f(s[2][r] - mnew);
      float p3 = __builtin_amdgcn_exp2f(s[3][r] - mnew);
      // C/D layout: VGPR r = row r (lanes 0-15) / r+8 (lanes 16-31), col = u*16+m
      int row = r + 8 * hi;
      __bf16* pr = pmine + row * 64 + m;
      pr[0]  = (__bf16)p0;
      pr[16] = (__bf16)p1;
      pr[32] = (__bf16)p2;
      pr[48] = (__bf16)p3;
      float ps = rowsum16((p0 + p1) + (p2 + p3));
      lrun[r] = lrun[r] * a + ps;
      mrun[r] = mnew;
      alpha[r] = a;
    }

    // rescale accumulators (alpha matches the acc row layout)
#pragma unroll
    for (int t = 0; t < 8; ++t)
#pragma unroll
      for (int r = 0; r < 8; ++r) acc[t][r] *= alpha[r];

    asm volatile("" ::: "memory");   // keep LDS P stores before A-frag reads

    // re-read P as two 16x32 A-fragments (wave-private LDS, same-wave order)
    const __bf16* prow = pmine + m * 64;
    FragA pf0, pf1;
    pf0.v = *reinterpret_cast<const v16bf*>(prow + hi * 16);
    pf1.v = *reinterpret_cast<const v16bf*>(prow + 32 + hi * 16);

    // ---- O += P x V over 8 channel tiles, 2 K-steps (16 WMMAs) ----
#pragma unroll
    for (int t = 0; t < 8; ++t) {
      const __bf16* vc = vcol0 + (size_t)t * 16 * Nn + j0 + hi * 16;
      v16bf bv0 = *reinterpret_cast<const v16bf*>(vc);
      acc[t] = __builtin_amdgcn_wmma_f32_16x16x32_bf16(false, pf0.v, false, bv0,
                                                       (short)0, acc[t], false, false);
      v16bf bv1 = *reinterpret_cast<const v16bf*>(vc + 32);
      acc[t] = __builtin_amdgcn_wmma_f32_16x16x32_bf16(false, pf1.v, false, bv1,
                                                       (short)0, acc[t], false, false);
    }
  }

  // epilogue: /rowsum, transpose through LDS, coalesced gamma*o + x
  float linv[8];
#pragma unroll
  for (int r = 0; r < 8; ++r) linv[r] = 1.0f / lrun[r];
#pragma unroll
  for (int t = 0; t < 8; ++t)
#pragma unroll
    for (int r = 0; r < 8; ++r) {
      int row_i = wi * 16 + r + 8 * hi;
      int cl    = ch * 128 + t * 16 + m;
      ot[cl * OTP + row_i] = acc[t][r] * linv[r];
    }
  __syncthreads();

  float g = gamma[0];
  for (int e = tid; e < Cc * 64; e += 256) {
    int c = e >> 6, i = e & 63;
    size_t gidx = ((size_t)(b * Cc + c)) * Nn + i0 + i;
    out[gidx] = g * ot[c * OTP + i] + x[gidx];
  }
}

// ---------------------------------------------------------------------------
extern "C" void kernel_launch(void* const* d_in, const int* in_sizes, int n_in,
                              void* d_out, int out_size, void* d_ws, size_t ws_size,
                              hipStream_t stream) {
  (void)in_sizes; (void)n_in; (void)out_size; (void)ws_size;
  const float* x     = (const float*)d_in[0];
  const float* Wq    = (const float*)d_in[1];
  const float* bq    = (const float*)d_in[2];
  const float* Wk    = (const float*)d_in[3];
  const float* bk    = (const float*)d_in[4];
  const float* Wv    = (const float*)d_in[5];
  const float* bv    = (const float*)d_in[6];
  const float* gamma = (const float*)d_in[7];
  float* out = (float*)d_out;

  // workspace: qT (2 MB) | kT (2 MB) | vB (16.8 MB), all bf16
  __bf16* qT = (__bf16*)d_ws;
  __bf16* kT = qT + (size_t)Bb * Nn * C8;
  __bf16* vB = kT + (size_t)Bb * Nn * C8;

  dim3 blk(256);
  dim3 g1(Nn / 32, Bb);
  qkv_bf16_kernel<<<g1, blk, 0, stream>>>(x, Wq, bq, Wk, bk, Wv, bv, qT, kT, vB);

  dim3 g2(Nn / 64, Bb);
  attn_flash_kernel<<<g2, blk, 0, stream>>>(qT, kT, vB, x, gamma, out);
}
